// DynamicEdgeConvLayer_18236431139303
// MI455X (gfx1250) — matrix operand order repaired and checked by reading, not compiled, not run on hardware
//
#include <hip/hip_runtime.h>

// Problem constants (match reference): B=8, C=64, N=4096, OUT=64, K=16
#define NPTS  4096
#define CCH   64
#define BATCH 8
#define KNN   16
#define OUTC  64

typedef __attribute__((ext_vector_type(16))) __bf16 v16bf;
typedef __attribute__((ext_vector_type(8)))  float  v8f;

struct alignas(16) B128 { unsigned int x[4]; };
struct alignas(16) F128 { float x[4]; };
union Frag { v16bf v; unsigned short u[16]; B128 q[2]; };

// zero-cost compiler fence: same-wave DS ops are executed in order by the LDS
// unit (ISA ch.5 / ch.11), so single-wave workgroups need no s_barrier and no
// counter drain -- only protection against compiler reordering.
#define WAVE_LDS_FENCE() asm volatile("" ::: "memory")

__device__ __forceinline__ unsigned short f2bf(float f) {
  unsigned u = __float_as_uint(f);
  unsigned r = (u + 0x7FFFu + ((u >> 16) & 1u)) >> 16;  // RNE
  return (unsigned short)r;
}
__device__ __forceinline__ float bf2f(unsigned short h) {
  return __uint_as_float(((unsigned)h) << 16);
}

// ---------------------------------------------------------------------------
// Kernel 0: x (B,C,N) f32 -> xb (B,N,C) bf16 ; sq[b][n] = sum_c x^2
// ---------------------------------------------------------------------------
__global__ __launch_bounds__(256) void prep_x(const float* __restrict__ x,
                                              unsigned short* __restrict__ xb,
                                              float* __restrict__ sq) {
  const int b = blockIdx.y;
  const int nbase = blockIdx.x * 64;
  const int tid = threadIdx.x;
  __shared__ float tile[64][65];  // [n][c], padded vs bank conflicts

  #pragma unroll
  for (int c0 = 0; c0 < 64; c0 += 4) {
    int c = c0 + (tid >> 6);
    int n = tid & 63;
    tile[n][c] = x[((size_t)(b * CCH + c)) * NPTS + nbase + n];  // coalesced over n
  }
  __syncthreads();   // 8 waves: real barrier required

  {
    int n = tid >> 2, g = tid & 3;
    unsigned short tmp[16];
    #pragma unroll
    for (int j = 0; j < 16; j++) tmp[j] = f2bf(tile[n][g * 16 + j]);
    B128 o0, o1;
    #pragma unroll
    for (int w = 0; w < 4; w++) {
      o0.x[w] = (unsigned)tmp[2 * w]     | ((unsigned)tmp[2 * w + 1] << 16);
      o1.x[w] = (unsigned)tmp[8 + 2 * w] | ((unsigned)tmp[8 + 2 * w + 1] << 16);
    }
    B128* dst = (B128*)(xb + ((size_t)(b * NPTS) + nbase + n) * CCH + g * 16);
    dst[0] = o0;
    dst[1] = o1;
  }

  if (tid < 64) {
    float s = 0.f;
    #pragma unroll
    for (int c = 0; c < 64; c++) { float v = tile[tid][c]; s += v * v; }
    sq[(size_t)b * NPTS + nbase + tid] = s;
  }
}

// ---------------------------------------------------------------------------
// Kernel 0b: weight transposes -> column-major bf16 fragments.
// W1 (128,64) -> W1t (64,128); W2 (64,64) -> W2t (64,64)
// ---------------------------------------------------------------------------
__global__ __launch_bounds__(256) void prep_w(const float* __restrict__ W1,
                                              const float* __restrict__ W2,
                                              unsigned short* __restrict__ W1t,
                                              unsigned short* __restrict__ W2t) {
  int t = threadIdx.x;
  for (int i = t; i < 128 * 64; i += 256) {
    int k = i >> 6, n = i & 63;
    W1t[n * 128 + k] = f2bf(W1[i]);
  }
  for (int i = t; i < 64 * 64; i += 256) {
    int k = i >> 6, n = i & 63;
    W2t[n * 64 + k] = f2bf(W2[i]);
  }
}

// ---------------------------------------------------------------------------
// knn helpers
// ---------------------------------------------------------------------------
struct BTiles { Frag t0k0, t0k1, t1k0, t1k1; };  // 2 candidate tiles x K halves
struct SVr    { F128 f[4]; };                    // 16 candidate sq-norms

__device__ __forceinline__ BTiles load_btiles(const unsigned short* __restrict__ xbB,
                                              int j, int m, int hi) {
  // B (32x16 bf16), contiguous k order used here and in A: lane<16 = column N=lane with K=0..15 contiguous,
  // lanes>=16 = K=16..31.
  BTiles r;
  const unsigned short* r0 = xbB + (size_t)(j + m) * CCH;
  const unsigned short* r1 = xbB + (size_t)(j + 16 + m) * CCH;
  const int k0 = hi * 16;
  r.t0k0.q[0] = *(const B128*)(r0 + k0);
  r.t0k0.q[1] = *(const B128*)(r0 + k0 + 8);
  r.t0k1.q[0] = *(const B128*)(r0 + 32 + k0);
  r.t0k1.q[1] = *(const B128*)(r0 + 32 + k0 + 8);
  r.t1k0.q[0] = *(const B128*)(r1 + k0);
  r.t1k0.q[1] = *(const B128*)(r1 + k0 + 8);
  r.t1k1.q[0] = *(const B128*)(r1 + 32 + k0);
  r.t1k1.q[1] = *(const B128*)(r1 + 32 + k0 + 8);
  return r;
}

__device__ __forceinline__ SVr load_sv(const float* __restrict__ sqB, int j, int hi) {
  // 16 sq values per half-wave, identical addresses across the half -> L1 bcast
  SVr r;
  const float* p = sqB + j + hi * 16;
  #pragma unroll
  for (int w = 0; w < 4; w++) r.f[w] = *(const F128*)(p + w * 4);
  return r;
}

__device__ __forceinline__ void topk_insert(float d, int id,
                                            float* bd, int* bi,
                                            float& cmax, int& carg) {
  if (d < cmax) {                 // rare after warm-up: 1 compare common path
    #pragma unroll
    for (int t = 0; t < 16; t++)
      if (t == carg) { bd[t] = d; bi[t] = id; }
    cmax = -3.4e38f;
    #pragma unroll
    for (int t = 0; t < 16; t++)
      if (bd[t] > cmax) { cmax = bd[t]; carg = t; }
  }
}

// ---------------------------------------------------------------------------
// Kernel 1: fused pairwise-distance + top-16.  One wave per 16 query rows.
// Per iteration: 2 candidate tiles (4 chained v_wmma_f32_16x16x32_bf16),
// software-pipelined B-fragment + sq loads (overlap next 32 candidates with
// WMMA + insertion), all 32 lanes do top-k (half-lists merged at the end).
// Score = sq[n] - 2*dot (sq[m] constant per row, dropped).
// Single-wave workgroup: no s_barrier / loadcnt drains, only compiler fences
// (same-wave DS ops are in-order).
// ---------------------------------------------------------------------------
__global__ __launch_bounds__(32) void knn_kernel(const unsigned short* __restrict__ xb,
                                                 const float* __restrict__ sq,
                                                 int* __restrict__ idx) {
  const int b = blockIdx.y;
  const int qbase = blockIdx.x * 16;
  const int lane = threadIdx.x;
  const int m = lane & 15;
  const int hi = lane >> 4;

  __shared__ float dot_lds[2 * 256];  // two 16x16 dot tiles, row-major
  __shared__ float mergeD[32 * 16];
  __shared__ int   mergeI[32 * 16];

  const unsigned short* xbB = xb + (size_t)b * NPTS * CCH;
  const float* sqB = sq + (size_t)b * NPTS;

  // A fragment (16x64 queries, 2 frags of 16x32) — resident all loop long.
  Frag a0, a1;
  {
    const unsigned short* qrow = xbB + (size_t)(qbase + m) * CCH;
    a0.q[0] = *(const B128*)(qrow + hi * 16);
    a0.q[1] = *(const B128*)(qrow + 8 + hi * 16);
    a1.q[0] = *(const B128*)(qrow + 32 + hi * 16);
    a1.q[1] = *(const B128*)(qrow + 40 + hi * 16);
  }

  float best_d[16];
  int   best_i[16];
  #pragma unroll
  for (int t = 0; t < 16; t++) { best_d[t] = 3.4e38f; best_i[t] = 0; }
  float cur_max = 3.4e38f;
  int   cur_arg = 0;

  BTiles cur = load_btiles(xbB, 0, m, hi);
  SVr    svc = load_sv(sqB, 0, hi);

  #pragma unroll 2   // even unroll -> SSA kills the buffer rotation copies
  for (int j = 0; j < NPTS; j += 32) {
    const int jn = (j + 32) & (NPTS - 1);
    BTiles nxt = load_btiles(xbB, jn, m, hi);     // prefetch next 32 candidates
    SVr    svn = load_sv(sqB, jn, hi);

    v8f acc0 = {}, acc1 = {};
    acc0 = __builtin_amdgcn_wmma_f32_16x16x32_bf16(false, a0.v, false, cur.t0k0.v,
                                                   (short)0, acc0, false, false);
    acc0 = __builtin_amdgcn_wmma_f32_16x16x32_bf16(false, a1.v, false, cur.t0k1.v,
                                                   (short)0, acc0, false, false);
    acc1 = __builtin_amdgcn_wmma_f32_16x16x32_bf16(false, a0.v, false, cur.t1k0.v,
                                                   (short)0, acc1, false, false);
    acc1 = __builtin_amdgcn_wmma_f32_16x16x32_bf16(false, a1.v, false, cur.t1k1.v,
                                                   (short)0, acc1, false, false);

    // C/D layout: VGPR r -> row (r + hi*8), col = lane&15
    #pragma unroll
    for (int r = 0; r < 8; r++) {
      dot_lds[(r + hi * 8) * 16 + m]       = acc0[r];
      dot_lds[256 + (r + hi * 8) * 16 + m] = acc1[r];
    }
    WAVE_LDS_FENCE();   // stores before loads; HW keeps same-wave DS in order

    // lane (hi,m): row m of tile `hi` -> vectorized b128 LDS reads
    float dv[16];
    #pragma unroll
    for (int w = 0; w < 4; w++)
      *(F128*)(&dv[w * 4]) = *(const F128*)(&dot_lds[hi * 256 + m * 16 + w * 4]);

    const int cb = j + hi * 16;
    #pragma unroll
    for (int nn = 0; nn < 16; nn++)
      topk_insert(svc.f[nn >> 2].x[nn & 3] - 2.0f * dv[nn], cb + nn,
                  best_d, best_i, cur_max, cur_arg);
    WAVE_LDS_FENCE();   // this iter's loads before next iter's stores

    cur = nxt;
    svc = svn;
  }

  // merge the two half-lists of each row (lane m & lane m+16) through LDS
  #pragma unroll
  for (int t = 0; t < 16; t++) {
    mergeD[lane * 16 + t] = best_d[t];
    mergeI[lane * 16 + t] = best_i[t];
  }
  WAVE_LDS_FENCE();
  if (lane < 16) {
    #pragma unroll
    for (int t = 0; t < 16; t++)
      topk_insert(mergeD[(lane + 16) * 16 + t], mergeI[(lane + 16) * 16 + t],
                  best_d, best_i, cur_max, cur_arg);
    int* o = idx + ((size_t)(b * NPTS) + qbase + lane) * KNN;
    #pragma unroll
    for (int t = 0; t < 16; t++) o[t] = best_i[t];  // order irrelevant (max over K)
  }
}

// ---------------------------------------------------------------------------
// Kernel 2: edge MLP + max-over-K.  One wave per point: M=16 (neighbors),
// layer1 16x128 @ 128x64 (16 WMMAs), layer2 16x64 @ 64x64 (8 WMMAs),
// then rowmax via VGPR max + ds_swizzle SWAPX16, relu(max + b2).
// ---------------------------------------------------------------------------
__global__ __launch_bounds__(32) void edge_mlp(const unsigned short* __restrict__ xb,
                                               const int* __restrict__ idx,
                                               const unsigned short* __restrict__ W1t,
                                               const unsigned short* __restrict__ W2t,
                                               const float* __restrict__ b1,
                                               const float* __restrict__ b2,
                                               float* __restrict__ out) {
  const int b = blockIdx.y;
  const int n = blockIdx.x;
  const int lane = threadIdx.x;
  const int kn = lane & 15;   // neighbor slot (= A-matrix row)
  const int hi = lane >> 4;

  __shared__ unsigned short h1[16 * 64];  // layer-1 activations, bf16, [k][feat]

  const int nbr = idx[((size_t)(b * NPTS) + n) * KNN + kn];
  const unsigned short* xi = xb + ((size_t)(b * NPTS) + n) * CCH;
  const unsigned short* xj = xb + ((size_t)(b * NPTS) + nbr) * CCH;

  // Build e = [xi (broadcast), xj - xi] as 4 A-fragments (K' = 0..127)
  Frag A[4];
  #pragma unroll
  for (int t = 0; t < 2; t++) {  // features 0..63: xi, identical across rows
    int c0 = 32 * t + hi * 8;
    A[t].q[0] = *(const B128*)(xi + c0);
    A[t].q[1] = *(const B128*)(xi + c0 + 16);
  }
  #pragma unroll
  for (int t = 2; t < 4; t++) {  // features 64..127: xj - xi
    int f0 = 32 * (t - 2) + hi * 8;
    #pragma unroll
    for (int half = 0; half < 2; half++) {
      int f = f0 + half * 16;
      Frag tj, ti;
      tj.q[0] = *(const B128*)(xj + f);
      ti.q[0] = *(const B128*)(xi + f);
      #pragma unroll
      for (int e = 0; e < 8; e++)
        A[t].u[half * 8 + e] = f2bf(bf2f(tj.u[e]) - bf2f(ti.u[e]));
    }
  }

  // Layer 1: relu(e @ W1 + b1) -> LDS (bf16)
  #pragma unroll
  for (int nt = 0; nt < 4; nt++) {
    v8f acc = {};
    #pragma unroll
    for (int t = 0; t < 4; t++) {
      Frag Bf;
      const unsigned short* wrow = W1t + (size_t)(nt * 16 + kn) * 128 + 32 * t + hi * 8;
      Bf.q[0] = *(const B128*)(wrow);
      Bf.q[1] = *(const B128*)(wrow + 16);
      acc = __builtin_amdgcn_wmma_f32_16x16x32_bf16(false, A[t].v, false, Bf.v,
                                                    (short)0, acc, false, false);
    }
    const float bias = b1[nt * 16 + kn];
    #pragma unroll
    for (int r = 0; r < 8; r++) {
      float v = acc[r] + bias;
      v = v > 0.f ? v : 0.f;
      h1[(r + hi * 8) * 64 + nt * 16 + kn] = f2bf(v);
    }
  }
  WAVE_LDS_FENCE();   // single wave: same-wave DS order suffices

  // Layer 2 A-fragments from LDS
  Frag A2[2];
  #pragma unroll
  for (int t = 0; t < 2; t++) {
    int c0 = 32 * t + hi * 8;
    A2[t].q[0] = *(const B128*)(&h1[kn * 64 + c0]);
    A2[t].q[1] = *(const B128*)(&h1[kn * 64 + c0 + 16]);
  }

  #pragma unroll
  for (int nt = 0; nt < 4; nt++) {
    v8f acc = {};
    #pragma unroll
    for (int t = 0; t < 2; t++) {
      Frag Bf;
      const unsigned short* wrow = W2t + (size_t)(nt * 16 + kn) * 64 + 32 * t + hi * 8;
      Bf.q[0] = *(const B128*)(wrow);
      Bf.q[1] = *(const B128*)(wrow + 16);
      acc = __builtin_amdgcn_wmma_f32_16x16x32_bf16(false, A2[t].v, false, Bf.v,
                                                    (short)0, acc, false, false);
    }
    // max over K: 8 rows in-lane, then SWAPX16 combines the two lane halves
    float mx = acc[0];
    #pragma unroll
    for (int r = 1; r < 8; r++) mx = fmaxf(mx, acc[r]);
    mx = fmaxf(mx, __int_as_float(
                       __builtin_amdgcn_ds_swizzle(__float_as_int(mx), 0x401F)));
    float v = mx + b2[nt * 16 + kn];  // relu(max(h)+b) == max(relu(h+b)) per column
    v = v > 0.f ? v : 0.f;
    if (hi == 0)
      out[((size_t)b * OUTC + nt * 16 + kn) * NPTS + n] = v;
  }
}

// ---------------------------------------------------------------------------
extern "C" void kernel_launch(void* const* d_in, const int* in_sizes, int n_in,
                              void* d_out, int out_size, void* d_ws, size_t ws_size,
                              hipStream_t stream) {
  (void)in_sizes; (void)n_in; (void)out_size; (void)ws_size;
  const float* x  = (const float*)d_in[0];
  const float* W1 = (const float*)d_in[1];
  const float* b1 = (const float*)d_in[2];
  const float* W2 = (const float*)d_in[3];
  const float* b2 = (const float*)d_in[4];
  float* out = (float*)d_out;

  // workspace layout (all 16B-aligned)
  char* ws = (char*)d_ws;
  unsigned short* xb  = (unsigned short*)(ws);             // 8*4096*64*2  = 4,194,304
  float*          sq  = (float*)(ws + 4194304);            // 8*4096*4    =   131,072
  int*            idx = (int*)(ws + 4325376);              // 8*4096*16*4 = 2,097,152
  unsigned short* W1t = (unsigned short*)(ws + 6422528);   // 64*128*2    =    16,384
  unsigned short* W2t = (unsigned short*)(ws + 6438912);   // 64*64*2     =     8,192

  prep_x  <<<dim3(NPTS / 64, BATCH), 256, 0, stream>>>(x, xb, sq);
  prep_w  <<<dim3(1, 1), 256, 0, stream>>>(W1, W2, W1t, W2t);
  knn_kernel<<<dim3(NPTS / 16, BATCH), 32, 0, stream>>>(xb, sq, idx);
  edge_mlp<<<dim3(NPTS, BATCH), 32, 0, stream>>>(xb, idx, W1t, W2t, b1, b2, out);
}
